// FasterRCNN_2671469658449
// MI455X (gfx1250) — matrix operand order repaired and checked by reading, not compiled, not run on hardware
//
#include <hip/hip_runtime.h>

// ---------------------------------------------------------------------------
// Faster-RCNN forward for MI455X (gfx1250, wave32, WMMA).
// All dense math goes through v_wmma_f32_16x16x32_bf16.
// ---------------------------------------------------------------------------

typedef __attribute__((ext_vector_type(16))) __bf16 v16bf;
typedef __attribute__((ext_vector_type(8)))  float  v8f;

#define BATCH   2
#define IMH     1024
#define IMW     1024
#define STRIDE  32
#define HFM     32            // H / STRIDE
#define WFM     32            // W / STRIDE
#define CFM     2048
#define NPOS    (BATCH*HFM*WFM)     // 2048 GEMM rows (b,y,x)
#define K_BB    (3*STRIDE*STRIDE)   // 3072
#define RPN_C   512
#define K_RPN   (9*CFM)             // 18432
#define NA      9
#define NANCH   (HFM*WFM*NA)        // 9216 per batch
#define PRE     2000
#define TOPN    1000
#define POOL    7
#define FCN     256
#define NCLS    21
#define ROIF    (CFM*POOL*POOL)     // 100352
#define NMS_TH  0.7f
#define CHUNK   250                 // rois per FC1 chunk (divides 1000)

// ---------------------------------------------------------------------------
// Generic bf16 WMMA GEMM:  C(MxN) = relu?(A(MxK) * B(KxN) + bias)
// A row-major bf16, B row-major (K-major) bf16. K must be a multiple of 64.
// Block: 256 threads = 8 waves laid out 4(rows) x 2(cols) of 16x16 tiles
// -> 64x32 macro tile, K stepped by 64 (2 WMMAs per barrier pair).
// A panel staged via 16B vector loads; fragments gathered per the CDNA5
// 16-bit A(16x32)/B(32x16) VGPR layouts (lowers to ds_load_b128).
// ---------------------------------------------------------------------------
__global__ __launch_bounds__(256) void wmma_gemm_bf16(
    const __bf16* __restrict__ Am, const __bf16* __restrict__ Bm,
    const float* __restrict__ bias, float* __restrict__ Cf,
    __bf16* __restrict__ Cb, int M, int N, int K, int relu)
{
    __shared__ __bf16 lA[64][72];    // 64 rows x 64 k (+8 pad, 144B pitch)
    __shared__ __bf16 lBt[32][80];   // 32 n   x 64 k (+16 pad, 160B pitch), transposed

    const int tid  = threadIdx.x;
    const int lane = tid & 31;
    const int wave = tid >> 5;
    const int wr   = wave >> 1;          // 0..3  (tile row)
    const int wc   = wave & 1;           // 0..1  (tile col)
    const int hv   = lane >> 4;          // lane half
    const int l15  = lane & 15;

    const int m0 = blockIdx.y * 64;
    const int n0 = blockIdx.x * 32;

    // cooperative loader indices
    const int arow = tid >> 2;           // 0..63
    const int aseg = (tid & 3) * 16;     // 0,16,32,48 (bf16 elems)
    const int bn   = tid & 31;           // 0..31 (n)
    const int bk   = (tid >> 5) * 8;     // 0..56 (k)
    const bool arOK = (m0 + arow) < M;
    const bool bnOK = (n0 + bn) < N;

    v8f acc = {0.f,0.f,0.f,0.f,0.f,0.f,0.f,0.f};

    for (int k0 = 0; k0 < K; k0 += 64) {
        // ---- stage A panel (64 x 64) : two 16B vector copies per thread ----
        {
            const __bf16* src = Am + (size_t)(m0 + arow) * K + (k0 + aseg);
            uint4 v0 = make_uint4(0u, 0u, 0u, 0u), v1 = v0;
            if (arOK) {
                v0 = *(const uint4*)(src);
                v1 = *(const uint4*)(src + 8);
            }
            *(uint4*)&lA[arow][aseg]     = v0;
            *(uint4*)&lA[arow][aseg + 8] = v1;
        }
        // ---- stage B panel transposed: lBt[n][k], 8 k per thread ----
        {
            __bf16 tmp[8];
            if (bnOK) {
                const __bf16* src = Bm + (size_t)(k0 + bk) * N + (n0 + bn);
                #pragma unroll
                for (int i = 0; i < 8; ++i) tmp[i] = src[(size_t)i * N];
            } else {
                #pragma unroll
                for (int i = 0; i < 8; ++i) tmp[i] = (__bf16)0.0f;
            }
            *(uint4*)&lBt[bn][bk] = *(const uint4*)tmp;
        }
        // prefetch next K tile (speculative; emits global_prefetch_b8)
        if (k0 + 64 < K) {
            __builtin_prefetch(Am + (size_t)(m0 + arow) * K + (k0 + 64 + aseg), 0, 3);
            __builtin_prefetch(Bm + (size_t)(k0 + 64 + bk) * N + (n0 + bn), 0, 3);
        }
        __syncthreads();

        // ---- two 16x16x32 WMMA sub-steps ----
        const __bf16* ar = &lA[wr * 16 + l15][0];
        const __bf16* br = &lBt[wc * 16 + l15][0];
        #pragma unroll
        for (int s = 0; s < 2; ++s) {
            // A 16x32: lanes 0-15 m=0..15; elems 0..7 -> k=0..7(+8*half),
            //          elems 8..15 -> k=16..23(+8*half)
            v16bf afrag, bfrag;
            const int ka = 32 * s + 8 * hv;
            #pragma unroll
            for (int i = 0; i < 8; ++i) afrag[i] = ar[ka + i];
            #pragma unroll
            for (int i = 0; i < 8; ++i) afrag[8 + i] = ar[16 + ka + i];
            // B 32x16: n=lane&15; elems 0..7 -> k=0..7(+8*half), elems 8..15 -> k=16..23(+8*half)
            const int kb = 32 * s + 8 * hv;
            #pragma unroll
            for (int i = 0; i < 8; ++i) bfrag[i] = br[kb + i];
            #pragma unroll
            for (int i = 0; i < 8; ++i) bfrag[8 + i] = br[16 + kb + i];
            acc = __builtin_amdgcn_wmma_f32_16x16x32_bf16(
                      false, afrag, false, bfrag, (short)0, acc, false, false);
        }
        __syncthreads();
    }

    // Epilogue. C/D layout: VGPR v -> m = v + 8*half, n = lane&15.
    const int n = n0 + wc * 16 + l15;
    #pragma unroll
    for (int v = 0; v < 8; ++v) {
        const int m = m0 + wr * 16 + hv * 8 + v;
        if (m < M && n < N) {
            float val = acc[v];
            if (bias) val += bias[n];
            if (relu) val = fmaxf(val, 0.0f);
            if (Cf) Cf[(size_t)m * N + n] = val;
            if (Cb) Cb[(size_t)m * N + n] = (__bf16)val;
        }
    }
}

// ---------------------------------------------------------------------------
// Packing / conversion kernels
// ---------------------------------------------------------------------------

// backbone im2col: non-overlapping 32x32 patches -> A1[r=b*1024+oy*32+ox][col=c*1024+iy*32+ix]
__global__ void pack_patches(const float* __restrict__ x, __bf16* __restrict__ A1)
{
    long long idx = (long long)blockIdx.x * 256 + threadIdx.x;
    if (idx >= (long long)NPOS * K_BB) return;
    int r = (int)(idx / K_BB), col = (int)(idx % K_BB);
    int b = r >> 10, pp = r & 1023;
    int oy = pp >> 5, ox = pp & 31;
    int c = col >> 10, rem = col & 1023;
    int iy = rem >> 5, ix = rem & 31;
    A1[idx] = (__bf16)x[(((size_t)(b * 3 + c) * IMH) + oy * 32 + iy) * IMW + ox * 32 + ix];
}

// generic (N,K) f32 -> (K,N) bf16 transpose (weights OIHW-flat -> K x N)
__global__ void transpose_nk_to_kn(const float* __restrict__ src,
                                   __bf16* __restrict__ dst, int N, int K)
{
    long long idx = (long long)blockIdx.x * 256 + threadIdx.x;
    if (idx >= (long long)N * K) return;
    int k = (int)(idx / N), n = (int)(idx % N);
    dst[idx] = (__bf16)src[(size_t)n * K + k];
}

// rpn weights (512,2048,3,3) -> B2[col=(dy*3+dx)*2048+c][oc]
__global__ void rpn_w_to_kn(const float* __restrict__ src, __bf16* __restrict__ dst)
{
    long long idx = (long long)blockIdx.x * 256 + threadIdx.x;
    if (idx >= (long long)K_RPN * RPN_C) return;
    int col = (int)(idx / RPN_C), oc = (int)(idx % RPN_C);
    int s = col / CFM, c = col % CFM;     // s = dy*3+dx
    dst[idx] = (__bf16)src[((size_t)oc * CFM + c) * 9 + s];
}

// rpn SAME-3x3 im2col from NHWC bf16 feature map
__global__ void rpn_im2col(const __bf16* __restrict__ fmB, __bf16* __restrict__ A2)
{
    long long idx = (long long)blockIdx.x * 256 + threadIdx.x;
    if (idx >= (long long)NPOS * K_RPN) return;
    int r = (int)(idx / K_RPN), col = (int)(idx % K_RPN);
    int b = r >> 10, pp = r & 1023;
    int y = pp >> 5, xq = pp & 31;
    int s = col / CFM, c = col & (CFM - 1);
    int dy = s / 3, dx = s % 3;
    int yy = y + dy - 1, xx = xq + dx - 1;
    __bf16 v = (__bf16)0.0f;
    if (yy >= 0 && yy < HFM && xx >= 0 && xx < WFM)
        v = fmB[(((size_t)b * HFM + yy) * WFM + xx) * CFM + c];
    A2[idx] = v;
}

// combined rpn cls+reg 1x1 weights -> B3 (512 x 54) bf16 + bias3(54)
__global__ void make_b3(const float* __restrict__ clsw, const float* __restrict__ regw,
                        const float* __restrict__ clsb, const float* __restrict__ regb,
                        __bf16* __restrict__ B3, float* __restrict__ bias3)
{
    int idx = blockIdx.x * 256 + threadIdx.x;
    if (idx < RPN_C * 54) {
        int k = idx / 54, n = idx % 54;
        float v = (n < 18) ? clsw[n * RPN_C + k] : regw[(n - 18) * RPN_C + k];
        B3[idx] = (__bf16)v;
    }
    if (idx < 54) bias3[idx] = (idx < 18) ? clsb[idx] : regb[idx - 18];
}

// combined cls(256x21)+reg(256x4) -> Bcr (256 x 25) bf16 + bcr(25)
__global__ void make_bcr(const float* __restrict__ clsw, const float* __restrict__ regw,
                         const float* __restrict__ clsb, const float* __restrict__ regb,
                         __bf16* __restrict__ Bcr, float* __restrict__ bcr)
{
    int idx = blockIdx.x * 256 + threadIdx.x;
    if (idx < FCN * 25) {
        int k = idx / 25, n = idx % 25;
        float v = (n < NCLS) ? clsw[k * NCLS + n] : regw[k * 4 + (n - NCLS)];
        Bcr[idx] = (__bf16)v;
    }
    if (idx < 25) bcr[idx] = (idx < NCLS) ? clsb[idx] : regb[idx - NCLS];
}

__global__ void f32_to_bf16(const float* __restrict__ src, __bf16* __restrict__ dst,
                            long long n)
{
    long long idx = (long long)blockIdx.x * 256 + threadIdx.x;
    if (idx < n) dst[idx] = (__bf16)src[idx];
}

// ---------------------------------------------------------------------------
// RPN post-processing: softmax fg + anchor decode + clip
// sreg rows: r = b*1024 + y*32 + x, cols 0..17 cls logits (2a,2a+1), 18..53 reg (4a..)
// ---------------------------------------------------------------------------
__global__ void proposals_kernel(const float* __restrict__ sreg,
                                 float* __restrict__ fg, float* __restrict__ prop)
{
    int idx = blockIdx.x * 256 + threadIdx.x;
    if (idx >= BATCH * NANCH) return;
    int b = idx / NANCH, i = idx % NANCH;
    int a = i % NA, p = i / NA;          // p = y*32 + x
    int y = p >> 5, x = p & 31;
    const float* row = sreg + (size_t)(b * (HFM * WFM) + p) * 54;

    float l0 = row[2 * a], l1 = row[2 * a + 1];
    fg[idx] = 1.0f / (1.0f + expf(l0 - l1));       // softmax fg prob

    const float scales[3] = {128.0f, 256.0f, 512.0f};
    const float ratios[3] = {0.5f, 1.0f, 2.0f};
    float sq = sqrtf(ratios[a % 3]);
    float ws = scales[a / 3] / sq;
    float hs = scales[a / 3] * sq;
    float cx = (x + 0.5f) * STRIDE;
    float cy = (y + 0.5f) * STRIDE;

    float dx = row[18 + 4 * a + 0];
    float dy = row[18 + 4 * a + 1];
    float dw = fminf(fmaxf(row[18 + 4 * a + 2], -4.0f), 4.0f);
    float dh = fminf(fmaxf(row[18 + 4 * a + 3], -4.0f), 4.0f);
    float ncx = cx + dx * ws, ncy = cy + dy * hs;
    float nw = ws * expf(dw), nh = hs * expf(dh);

    float* q = prop + (size_t)idx * 4;
    q[0] = fminf(fmaxf(ncx - 0.5f * nw, 0.0f), (float)(IMW - 1));
    q[1] = fminf(fmaxf(ncy - 0.5f * nh, 0.0f), (float)(IMH - 1));
    q[2] = fminf(fmaxf(ncx + 0.5f * nw, 0.0f), (float)(IMW - 1));
    q[3] = fminf(fmaxf(ncy + 0.5f * nh, 0.0f), (float)(IMH - 1));
}

// ---------------------------------------------------------------------------
// Per-batch top-PRE selection (iterative LDS argmax, stable tie-break on idx)
// ---------------------------------------------------------------------------
__global__ __launch_bounds__(256) void topk_kernel(const float* __restrict__ fg,
                                                   const float* __restrict__ prop,
                                                   float* __restrict__ sortedB,
                                                   float* __restrict__ sortedS)
{
    int b = blockIdx.x;
    __shared__ float sc[NANCH];
    __shared__ float rmax[256];
    __shared__ int   ridx[256];
    for (int i = threadIdx.x; i < NANCH; i += 256) sc[i] = fg[(size_t)b * NANCH + i];
    __syncthreads();

    for (int it = 0; it < PRE; ++it) {
        float bv = -1e38f; int bi = 0x7fffffff;
        for (int i = threadIdx.x; i < NANCH; i += 256) {
            float v = sc[i];
            if (v > bv || (v == bv && i < bi)) { bv = v; bi = i; }
        }
        rmax[threadIdx.x] = bv; ridx[threadIdx.x] = bi;
        __syncthreads();
        for (int s = 128; s > 0; s >>= 1) {
            if (threadIdx.x < s) {
                float ov = rmax[threadIdx.x + s]; int oi = ridx[threadIdx.x + s];
                if (ov > rmax[threadIdx.x] ||
                    (ov == rmax[threadIdx.x] && oi < ridx[threadIdx.x])) {
                    rmax[threadIdx.x] = ov; ridx[threadIdx.x] = oi;
                }
            }
            __syncthreads();
        }
        if (threadIdx.x == 0) {
            int bidx = ridx[0];
            sortedS[(size_t)b * PRE + it] = rmax[0];
            const float* pp = prop + ((size_t)b * NANCH + bidx) * 4;
            float* q = sortedB + ((size_t)b * PRE + it) * 4;
            q[0] = pp[0]; q[1] = pp[1]; q[2] = pp[2]; q[3] = pp[3];
            sc[bidx] = -1e38f;
        }
        __syncthreads();
    }
}

// ---------------------------------------------------------------------------
// Per-batch NMS (serial over i, parallel over j) + compaction to TOPN
// ---------------------------------------------------------------------------
__global__ __launch_bounds__(256) void nms_kernel(const float* __restrict__ sortedB,
                                                  float* __restrict__ props,
                                                  int* __restrict__ valid)
{
    int b = blockIdx.x;
    __shared__ float bx[PRE][4];
    __shared__ unsigned char keep[PRE];
    for (int i = threadIdx.x; i < PRE; i += 256) {
        const float* p = sortedB + ((size_t)b * PRE + i) * 4;
        bx[i][0] = p[0]; bx[i][1] = p[1]; bx[i][2] = p[2]; bx[i][3] = p[3];
        keep[i] = 1;
    }
    __syncthreads();

    for (int i = 0; i < PRE - 1; ++i) {
        if (keep[i]) {
            float ax1 = bx[i][0], ay1 = bx[i][1], ax2 = bx[i][2], ay2 = bx[i][3];
            float areaA = (ax2 - ax1) * (ay2 - ay1);
            for (int j = i + 1 + threadIdx.x; j < PRE; j += 256) {
                float bx1 = bx[j][0], by1 = bx[j][1], bx2 = bx[j][2], by2 = bx[j][3];
                float areaB = (bx2 - bx1) * (by2 - by1);
                float lx = fmaxf(ax1, bx1), ly = fmaxf(ay1, by1);
                float rx = fminf(ax2, bx2), ry = fminf(ay2, by2);
                float iw = fmaxf(rx - lx, 0.0f), ih = fmaxf(ry - ly, 0.0f);
                float inter = iw * ih;
                float iou = inter / (areaA + areaB - inter + 1e-9f);
                if (iou > NMS_TH) keep[j] = 0;
            }
        }
        __syncthreads();
    }

    if (threadIdx.x == 0) {
        int cnt = 0;
        for (int i = 0; i < PRE && cnt < TOPN; ++i) {
            if (keep[i]) {
                float* q = props + ((size_t)b * TOPN + cnt) * 4;
                q[0] = bx[i][0]; q[1] = bx[i][1]; q[2] = bx[i][2]; q[3] = bx[i][3];
                valid[b * TOPN + cnt] = 1;
                ++cnt;
            }
        }
        for (; cnt < TOPN; ++cnt) {
            float* q = props + ((size_t)b * TOPN + cnt) * 4;
            q[0] = 0.f; q[1] = 0.f; q[2] = 0.f; q[3] = 0.f;
            valid[b * TOPN + cnt] = 0;
        }
    }
}

// ---------------------------------------------------------------------------
// ROI align fused with valid-mask; writes a CHUNK x ROIF bf16 A-matrix whose
// column order f = c*49 + py*7 + px matches fc1_w's flatten order.
// grid: (CHUNK, 49), block 256 over channels
// ---------------------------------------------------------------------------
__global__ __launch_bounds__(256) void roi_align_kernel(
    const float* __restrict__ fmf, const float* __restrict__ props,
    const int* __restrict__ valid, __bf16* __restrict__ Aroi, int b, int n0)
{
    int nl = blockIdx.x;
    int pp = blockIdx.y;
    int px = pp % POOL, py = pp / POOL;
    int gid = b * TOPN + n0 + nl;
    float vm = valid[gid] ? 1.0f : 0.0f;
    const float* roi = props + (size_t)gid * 4;

    const float sc = 1.0f / (float)STRIDE;
    float x1 = roi[0] * sc, y1 = roi[1] * sc, x2 = roi[2] * sc, y2 = roi[3] * sc;
    float xs = fminf(fmaxf(x1 + (px + 0.5f) * (x2 - x1) / POOL, 0.0f), (float)(WFM - 1));
    float ys = fminf(fmaxf(y1 + (py + 0.5f) * (y2 - y1) / POOL, 0.0f), (float)(HFM - 1));
    float x0f = floorf(xs), y0f = floorf(ys);
    int x0i = (int)x0f, y0i = (int)y0f;
    int x1i = min(x0i + 1, WFM - 1), y1i = min(y0i + 1, HFM - 1);
    float wx = xs - x0f, wy = ys - y0f;
    float w00 = (1.f - wy) * (1.f - wx), w01 = (1.f - wy) * wx;
    float w10 = wy * (1.f - wx),         w11 = wy * wx;

    const float* base = fmf + (size_t)b * HFM * WFM * CFM;
    const float* p00 = base + ((size_t)y0i * WFM + x0i) * CFM;
    const float* p01 = base + ((size_t)y0i * WFM + x1i) * CFM;
    const float* p10 = base + ((size_t)y1i * WFM + x0i) * CFM;
    const float* p11 = base + ((size_t)y1i * WFM + x1i) * CFM;

    for (int c = threadIdx.x; c < CFM; c += 256) {
        float v = w00 * p00[c] + w01 * p01[c] + w10 * p10[c] + w11 * p11[c];
        Aroi[(size_t)nl * ROIF + (size_t)c * (POOL * POOL) + pp] = (__bf16)(v * vm);
    }
}

// ---------------------------------------------------------------------------
// Final head: split cls/reg, apply valid mask, decode + clip detections.
// out layout: [0, 8000) detections (B,1000,4); [8000, 50000) cls (B,1000,21)
// ---------------------------------------------------------------------------
__global__ void final_kernel(const float* __restrict__ ocr,
                             const float* __restrict__ props,
                             const int* __restrict__ valid, float* __restrict__ out)
{
    int idx = blockIdx.x * 256 + threadIdx.x;
    if (idx >= BATCH * TOPN) return;
    float vm = valid[idx] ? 1.0f : 0.0f;
    const float* o = ocr + (size_t)idx * 25;

    float* co = out + BATCH * TOPN * 4 + (size_t)idx * NCLS;
    #pragma unroll
    for (int c = 0; c < NCLS; ++c) co[c] = o[c] * vm;

    float dx = o[21] * vm, dy = o[22] * vm;
    float dw = fminf(fmaxf(o[23] * vm, -4.0f), 4.0f);
    float dh = fminf(fmaxf(o[24] * vm, -4.0f), 4.0f);

    const float* pb = props + (size_t)idx * 4;
    float w = pb[2] - pb[0], h = pb[3] - pb[1];
    float cx = pb[0] + 0.5f * w, cy = pb[1] + 0.5f * h;
    float ncx = cx + dx * w, ncy = cy + dy * h;
    float nw = w * expf(dw), nh = h * expf(dh);

    float* dq = out + (size_t)idx * 4;
    dq[0] = fminf(fmaxf(ncx - 0.5f * nw, 0.0f), (float)(IMW - 1)) * vm;
    dq[1] = fminf(fmaxf(ncy - 0.5f * nh, 0.0f), (float)(IMH - 1)) * vm;
    dq[2] = fminf(fmaxf(ncx + 0.5f * nw, 0.0f), (float)(IMW - 1)) * vm;
    dq[3] = fminf(fmaxf(ncy + 0.5f * nh, 0.0f), (float)(IMH - 1)) * vm;
}

// ---------------------------------------------------------------------------
// Host orchestration
// ---------------------------------------------------------------------------
static inline int nblk(long long n) { return (int)((n + 255) / 256); }

extern "C" void kernel_launch(void* const* d_in, const int* in_sizes, int n_in,
                              void* d_out, int out_size, void* d_ws, size_t ws_size,
                              hipStream_t stream)
{
    (void)in_sizes; (void)n_in; (void)out_size; (void)ws_size;

    const float* x          = (const float*)d_in[0];
    const float* backbone_w = (const float*)d_in[1];
    const float* backbone_b = (const float*)d_in[2];
    const float* rpn_w      = (const float*)d_in[3];
    const float* rpn_b      = (const float*)d_in[4];
    const float* rpn_cls_w  = (const float*)d_in[5];
    const float* rpn_cls_b  = (const float*)d_in[6];
    const float* rpn_reg_w  = (const float*)d_in[7];
    const float* rpn_reg_b  = (const float*)d_in[8];
    const float* fc1_w      = (const float*)d_in[9];
    const float* fc1_b      = (const float*)d_in[10];
    const float* fc2_w      = (const float*)d_in[11];
    const float* fc2_b      = (const float*)d_in[12];
    const float* cls_w      = (const float*)d_in[13];
    const float* cls_b      = (const float*)d_in[14];
    const float* reg_w      = (const float*)d_in[15];
    const float* reg_b      = (const float*)d_in[16];
    float* out = (float*)d_out;
    char*  ws  = (char*)d_ws;

    // ---- scratch layout (manual, with lifetime reuse) ----
    const size_t offA1  = 0;                                    // 2048x3072 bf16
    const size_t offB1  = offA1 + (size_t)NPOS * K_BB * 2;      // 3072x2048 bf16
    const size_t offFmF = offB1 + (size_t)K_BB * CFM * 2;       // 2048x2048 f32 (NHWC)
    const size_t offFmB = offFmF + (size_t)NPOS * CFM * 4;      // bf16 copy
    const size_t offA2  = offFmB + (size_t)NPOS * CFM * 2;      // 2048x18432 bf16
    // A1/B1 dead after GEMM1 -> reuse for B2 + h
    const size_t offB2  = 0;                                    // 18432x512 bf16 (18.9MB < 25.2MB)
    const size_t offHb  = offB2 + (size_t)K_RPN * RPN_C * 2;    // 2048x512 bf16
    // A2 dead after GEMM2 -> bump allocate the rest from offA2
    size_t p = offA2;
    auto take = [&](size_t bytes) { size_t r = p; p += (bytes + 255) & ~(size_t)255; return r; };
    const size_t offB3   = take((size_t)RPN_C * 54 * 2);
    const size_t offBs3  = take(54 * 4);
    const size_t offSreg = take((size_t)NPOS * 54 * 4);
    const size_t offFg   = take((size_t)BATCH * NANCH * 4);
    const size_t offProp = take((size_t)BATCH * NANCH * 16);
    const size_t offSb   = take((size_t)BATCH * PRE * 16);
    const size_t offSs   = take((size_t)BATCH * PRE * 4);
    const size_t offPr   = take((size_t)BATCH * TOPN * 16);
    const size_t offVa   = take((size_t)BATCH * TOPN * 4);
    const size_t offBcr  = take((size_t)FCN * 25 * 2);
    const size_t offbcr  = take(25 * 4);
    const size_t offF1b  = take((size_t)ROIF * FCN * 2);        // fc1_w bf16 (51MB)
    const size_t offAroi = take((size_t)CHUNK * ROIF * 2);      // 50MB chunk
    const size_t offH1b  = take((size_t)BATCH * TOPN * FCN * 2);
    const size_t offF2b  = take((size_t)FCN * FCN * 2);
    const size_t offH2b  = take((size_t)BATCH * TOPN * FCN * 2);
    const size_t offOcr  = take((size_t)BATCH * TOPN * 25 * 4);

    __bf16* A1   = (__bf16*)(ws + offA1);
    __bf16* B1   = (__bf16*)(ws + offB1);
    float*  FmF  = (float*)(ws + offFmF);
    __bf16* FmB  = (__bf16*)(ws + offFmB);
    __bf16* A2   = (__bf16*)(ws + offA2);
    __bf16* B2   = (__bf16*)(ws + offB2);
    __bf16* Hb   = (__bf16*)(ws + offHb);
    __bf16* B3   = (__bf16*)(ws + offB3);
    float*  Bs3  = (float*)(ws + offBs3);
    float*  Sreg = (float*)(ws + offSreg);
    float*  Fg   = (float*)(ws + offFg);
    float*  Prop = (float*)(ws + offProp);
    float*  Sb   = (float*)(ws + offSb);
    float*  Ss   = (float*)(ws + offSs);
    float*  Pr   = (float*)(ws + offPr);
    int*    Va   = (int*)(ws + offVa);
    __bf16* Bcr  = (__bf16*)(ws + offBcr);
    float*  bcr  = (float*)(ws + offbcr);
    __bf16* F1b  = (__bf16*)(ws + offF1b);
    __bf16* Aroi = (__bf16*)(ws + offAroi);
    __bf16* H1b  = (__bf16*)(ws + offH1b);
    __bf16* F2b  = (__bf16*)(ws + offF2b);
    __bf16* H2b  = (__bf16*)(ws + offH2b);
    float*  Ocr  = (float*)(ws + offOcr);

    // 1) backbone conv as WMMA GEMM (M=2048,N=2048,K=3072), ReLU -> fm (f32+bf16, NHWC)
    pack_patches<<<nblk((long long)NPOS * K_BB), 256, 0, stream>>>(x, A1);
    transpose_nk_to_kn<<<nblk((long long)CFM * K_BB), 256, 0, stream>>>(backbone_w, B1, CFM, K_BB);
    wmma_gemm_bf16<<<dim3(CFM / 32, NPOS / 64), 256, 0, stream>>>(
        A1, B1, backbone_b, FmF, FmB, NPOS, CFM, K_BB, 1);

    // 2) RPN 3x3 conv as WMMA GEMM (M=2048,N=512,K=18432), ReLU
    rpn_im2col<<<nblk((long long)NPOS * K_RPN), 256, 0, stream>>>(FmB, A2);
    rpn_w_to_kn<<<nblk((long long)K_RPN * RPN_C), 256, 0, stream>>>(rpn_w, B2);
    wmma_gemm_bf16<<<dim3(RPN_C / 32, NPOS / 64), 256, 0, stream>>>(
        A2, B2, rpn_b, nullptr, Hb, NPOS, RPN_C, K_RPN, 1);

    // 3) RPN heads (cls 18 + reg 36 fused, N=54)
    make_b3<<<nblk(RPN_C * 54), 256, 0, stream>>>(rpn_cls_w, rpn_reg_w, rpn_cls_b, rpn_reg_b, B3, Bs3);
    wmma_gemm_bf16<<<dim3((54 + 31) / 32, NPOS / 64), 256, 0, stream>>>(
        Hb, B3, Bs3, Sreg, nullptr, NPOS, 54, RPN_C, 0);

    // 4) softmax + anchor decode + clip
    proposals_kernel<<<nblk(BATCH * NANCH), 256, 0, stream>>>(Sreg, Fg, Prop);

    // 5) per-batch top-2000 and NMS -> props/valid
    topk_kernel<<<BATCH, 256, 0, stream>>>(Fg, Prop, Sb, Ss);
    nms_kernel<<<BATCH, 256, 0, stream>>>(Sb, Pr, Va);

    // 6) weight conversions for FC stack
    f32_to_bf16<<<nblk((long long)ROIF * FCN), 256, 0, stream>>>(fc1_w, F1b, (long long)ROIF * FCN);
    f32_to_bf16<<<nblk((long long)FCN * FCN), 256, 0, stream>>>(fc2_w, F2b, (long long)FCN * FCN);
    make_bcr<<<nblk(FCN * 25), 256, 0, stream>>>(cls_w, reg_w, cls_b, reg_b, Bcr, bcr);

    // 7) ROI align + FC1 (M=250 chunks, N=256, K=100352), ReLU
    for (int b = 0; b < BATCH; ++b) {
        for (int n0 = 0; n0 < TOPN; n0 += CHUNK) {
            roi_align_kernel<<<dim3(CHUNK, POOL * POOL), 256, 0, stream>>>(
                FmF, Pr, Va, Aroi, b, n0);
            wmma_gemm_bf16<<<dim3(FCN / 32, (CHUNK + 63) / 64), 256, 0, stream>>>(
                Aroi, F1b, fc1_b, nullptr, H1b + (size_t)(b * TOPN + n0) * FCN,
                CHUNK, FCN, ROIF, 1);
        }
    }

    // 8) FC2 (M=2000,N=256,K=256), ReLU
    wmma_gemm_bf16<<<dim3(FCN / 32, (BATCH * TOPN + 63) / 64), 256, 0, stream>>>(
        H1b, F2b, fc2_b, nullptr, H2b, BATCH * TOPN, FCN, FCN, 1);

    // 9) cls+reg heads (N=25)
    wmma_gemm_bf16<<<dim3(1, (BATCH * TOPN + 63) / 64), 256, 0, stream>>>(
        H2b, Bcr, bcr, Ocr, nullptr, BATCH * TOPN, 25, FCN, 0);

    // 10) mask, decode, clip -> d_out (detections then cls_scores)
    final_kernel<<<nblk(BATCH * TOPN), 256, 0, stream>>>(Ocr, Pr, Va, out);
}
